// SPAM_62517543961204
// MI455X (gfx1250) — matrix operand order repaired and checked by reading, not compiled, not run on hardware
//
#include <hip/hip_runtime.h>
#include <hip/hip_bf16.h>

typedef __attribute__((ext_vector_type(16))) __bf16 bf16x16;
typedef __attribute__((ext_vector_type(8)))  __bf16 bf16x8;
typedef __attribute__((ext_vector_type(8)))  float  floatx8;

constexpr int Bsz = 4, C = 256, H = 128, W = 256, C0 = 64, Dd = 32, Mbr = 4;
constexpr int HW = H * W;

__device__ __forceinline__ floatx8 wmma_bf16(bf16x16 a, bf16x16 b, floatx8 c) {
  return __builtin_amdgcn_wmma_f32_16x16x32_bf16(false, a, false, b, (short)0, c, false, false);
}

// ---------------- prep kernels ----------------

__global__ void k_fold(const float* bng, const float* bnb, const float* bnm, const float* bnv,
                       const float* skg, const float* skb_, const float* skm, const float* skv,
                       const float* f1g, const float* f1b_, const float* f1m, const float* f1v,
                       float* BNS, float* BNB, float* SKS, float* SKB, float* F1S, float* F1B) {
  int t = threadIdx.x + blockIdx.x * blockDim.x;
  if (t < C)      { float s = bng[t] * rsqrtf(bnv[t] + 1e-5f); BNS[t] = s; BNB[t] = bnb[t] - bnm[t] * s; }
  if (t < Mbr*C)  { float s = skg[t] * rsqrtf(skv[t] + 1e-5f); SKS[t] = s; SKB[t] = skb_[t] - skm[t] * s; }
  if (t < Dd)     { float s = f1g[t] * rsqrtf(f1v[t] + 1e-5f); F1S[t] = s; F1B[t] = f1b_[t] - f1m[t] * s; }
}

__global__ void k_tobf(const float* __restrict__ in, __bf16* __restrict__ out, size_t n) {
  size_t i = (size_t)blockIdx.x * blockDim.x + threadIdx.x;
  if (i < n) out[i] = (__bf16)in[i];
}

// Pack conv weights into WMMA A-fragment order: [br][mt][pair p][lane][e]
__global__ void k_packw(const float* __restrict__ skw, __bf16* __restrict__ wp) {
  int lane = threadIdx.x;
  int z = blockIdx.x;                 // (br*16+mt)*5+p
  int p = z % 5, mt = (z / 5) & 15, br = z / 80;
  int m = lane & 15, kd = lane >> 4;
  __bf16* dst = wp + ((size_t)z * 32 + lane) * 16;
  for (int e = 0; e < 16; ++e) {
    int k = (kd << 4) | e;
    int tap = 2 * p + (k >> 4);
    int c16 = k & 15;
    float wv = 0.f;
    if (tap < 9 && ((c16 >> 3) == (m >> 3)))
      wv = skw[((size_t)(br * C + mt * 16 + m) * 8 + (c16 & 7)) * 9 + tap];
    dst[e] = (__bf16)wv;
  }
}

// Pack bq_w/bs_w into A-fragment order: [side][g][ks][lane][e]
__global__ void k_packqw(const float* __restrict__ bqw, const float* __restrict__ bsw,
                         __bf16* __restrict__ qwp) {
  int lane = threadIdx.x;
  int z = blockIdx.x;                 // side*8 + g*2 + ks
  int side = z >> 3, g = (z >> 1) & 3, ks = z & 1;
  const float* wsrc = side ? bsw : bqw;
  int m = lane & 15, kd = lane >> 4;
  __bf16* dst = qwp + ((size_t)z * 32 + lane) * 16;
  for (int e = 0; e < 16; ++e) {
    int ic = ks * 32 + ((kd << 4) | e);
    dst[e] = (__bf16)wsrc[(g * 16 + m) * 64 + ic];
  }
}

// BN(catfea) -> bf16 in channel-interleaved tiles: xb2[b][ct16][pixel][c16]
__global__ void k_bncat(const float* __restrict__ cat, const float* __restrict__ BNS,
                        const float* __restrict__ BNB, __bf16* __restrict__ xb2) {
  __shared__ __bf16 tile[16][17];
  int t = threadIdx.x;
  size_t p0 = (size_t)blockIdx.x * 16;
  int ct = blockIdx.y, b = blockIdx.z;
  int x = t & 15, c16 = t >> 4;
  int c = ct * 16 + c16;
  float v = cat[((size_t)(b * C + c)) * HW + p0 + x] * BNS[c] + BNB[c];
  tile[c16][x] = (__bf16)v;
  __syncthreads();
  int c2 = t & 15, x2 = t >> 4;
  xb2[(((size_t)(b * 16 + ct)) * HW + p0 + x2) * 16 + c2] = tile[c2][x2];
}

// ---------------- grouped dilated 3x3 conv via WMMA ----------------
// grid (W/16, H, B*4branches*16mtiles), block 32
__global__ void k_skconv(const __bf16* __restrict__ xb2, const __bf16* __restrict__ wp,
                         const float* __restrict__ sks, const float* __restrict__ skb,
                         __bf16* __restrict__ outs) {
  const int lane = threadIdx.x;
  const int x = blockIdx.x * 16 + (lane & 15);
  const int y = blockIdx.y;
  int z = blockIdx.z;
  const int b = z >> 6; z &= 63;
  const int br = z >> 4, mt = z & 15;
  const int dl = br + 1;
  const int kd = lane >> 4;
  const __bf16* xbase = xb2 + (size_t)(b * 16 + mt) * HW * 16;
  floatx8 acc = {};
#pragma unroll
  for (int p = 0; p < 5; ++p) {
    bf16x16 afrag = *(const bf16x16*)(wp + (((size_t)((br * 16 + mt) * 5 + p)) * 32 + lane) * 16);
    int tr = 2 * p + kd;
    int tap = tr > 8 ? 8 : tr;
    int ty = tap / 3, tx = tap % 3;
    int yy = y + (ty - 1) * dl;
    int xx = x + (tx - 1) * dl;
    bool valid = (tr <= 8) && (yy >= 0) && (yy < H) && (xx >= 0) && (xx < W);
    int yyc = yy < 0 ? 0 : (yy >= H ? H - 1 : yy);
    int xxc = xx < 0 ? 0 : (xx >= W ? W - 1 : xx);
    bf16x16 raw = *(const bf16x16*)(xbase + ((size_t)yyc * W + xxc) * 16);
    bf16x16 bfrag;
#pragma unroll
    for (int e = 0; e < 16; ++e) bfrag[e] = valid ? raw[e] : (__bf16)0.f;
    acc = wmma_bf16(afrag, bfrag, acc);
  }
  __bf16* op = outs + ((size_t)br * Bsz + b) * C * (size_t)HW;
#pragma unroll
  for (int r = 0; r < 8; ++r) {
    int co = mt * 16 + r + (kd << 3);
    float v = acc[r] * sks[br * C + co] + skb[br * C + co];
    v = v > 0.f ? v : 0.f;
    op[(size_t)co * HW + y * W + x] = (__bf16)v;
  }
}

// per-(b,c) sum of all 4 branch outputs
__global__ void k_pool(const __bf16* __restrict__ outs, float* __restrict__ pool) {
  __shared__ float red[256];
  int t = threadIdx.x;
  int bc = blockIdx.x;
  int b = bc / C, c = bc % C;
  float acc = 0.f;
  for (int i = 0; i < Mbr; ++i) {
    const __bf16* p = outs + (((size_t)i * Bsz + b) * C + c) * HW;
    for (int j = t; j < HW; j += 256) acc += (float)p[j];
  }
  red[t] = acc; __syncthreads();
  for (int s = 128; s > 0; s >>= 1) { if (t < s) red[t] += red[t + s]; __syncthreads(); }
  if (t == 0) pool[bc] = red[0];
}

// SE: fc1+bn+relu -> fc2 -> softmax over branches
__global__ void k_skattn(const float* __restrict__ pool, const float* __restrict__ fc1w,
                         const float* __restrict__ f1s, const float* __restrict__ f1b,
                         const float* __restrict__ fc2w, float* __restrict__ a) {
  __shared__ float zl[Bsz][Dd];
  int t = threadIdx.x;
  if (t < Bsz * Dd) {
    int b = t >> 5, j = t & 31;
    float acc = 0.f;
    for (int c = 0; c < C; ++c) acc += fc1w[j * C + c] * pool[b * C + c];
    acc *= (1.f / (float)HW);
    float zz = acc * f1s[j] + f1b[j];
    zl[b][j] = zz > 0.f ? zz : 0.f;
  }
  __syncthreads();
  for (int idx = t; idx < Bsz * C; idx += 256) {
    int b = idx / C, c = idx % C;
    float l[Mbr]; float mx = -1e30f;
    for (int i = 0; i < Mbr; ++i) {
      float acc = 0.f;
      for (int j = 0; j < Dd; ++j) acc += fc2w[((size_t)i * C + c) * Dd + j] * zl[b][j];
      l[i] = acc; mx = fmaxf(mx, acc);
    }
    float s = 0.f;
    for (int i = 0; i < Mbr; ++i) { l[i] = __expf(l[i] - mx); s += l[i]; }
    float inv = 1.f / s;
    for (int i = 0; i < Mbr; ++i) a[(b * Mbr + i) * C + c] = l[i] * inv;
  }
}

// V = sum_i a[b,i,c]*outs_i -> interleaved Vb[b][ct16][pixel][c16]
__global__ void k_wv(const __bf16* __restrict__ outs, const float* __restrict__ a,
                     __bf16* __restrict__ Vb) {
  __shared__ __bf16 tile[16][17];
  int t = threadIdx.x;
  size_t p0 = (size_t)blockIdx.x * 16;
  int ct = blockIdx.y, b = blockIdx.z;
  int x = t & 15, c16 = t >> 4;
  int c = ct * 16 + c16;
  float acc = 0.f;
#pragma unroll
  for (int br = 0; br < Mbr; ++br)
    acc += a[(b * Mbr + br) * C + c] * (float)outs[(((size_t)br * Bsz + b) * C + c) * HW + p0 + x];
  tile[c16][x] = (__bf16)acc;
  __syncthreads();
  int c2 = t & 15, x2 = t >> 4;
  Vb[(((size_t)(b * 16 + ct)) * HW + p0 + x2) * 16 + c2] = tile[c2][x2];
}

// grouped 1x1 conv (256->64, groups=4) via WMMA
__global__ void k_qconv(const __bf16* __restrict__ Vb, const __bf16* __restrict__ qwp,
                        const float* __restrict__ bq, float* __restrict__ Qraw, int side) {
  const int lane = threadIdx.x;
  const int x = blockIdx.x * 16 + (lane & 15);
  const int y = blockIdx.y;
  const int b = blockIdx.z >> 2, g = blockIdx.z & 3;
  const int kd = lane >> 4;
  floatx8 acc = {};
#pragma unroll
  for (int ks = 0; ks < 2; ++ks) {
    bf16x16 a = *(const bf16x16*)(qwp + (((size_t)(side * 8 + g * 2 + ks)) * 32 + lane) * 16);
    int ctile = b * 16 + g * 4 + ks * 2 + kd;
    bf16x16 bb = *(const bf16x16*)(Vb + ((size_t)ctile * HW + y * W + x) * 16);
    acc = wmma_bf16(a, bb, acc);
  }
#pragma unroll
  for (int r = 0; r < 8; ++r) {
    int co = g * 16 + r + (kd << 3);
    Qraw[((size_t)(b * C0 + co) * H + y) * W + x] = acc[r] + bq[co];
  }
}

// subtract row-mean over w; emit bf16 transposed to [b,h,w,c]
__global__ void k_meansub(const float* __restrict__ Qraw, __bf16* __restrict__ Qt) {
  __shared__ float buf[32][W];
  __shared__ float mean[32];
  int t = threadIdx.x;
  int half = blockIdx.y;
  size_t bh = blockIdx.x;
  int b = (int)(bh / H), h = (int)(bh % H);
  int c0 = half * 32;
  for (int i = t; i < 32 * W; i += 256) {
    int ci = i >> 8, w = i & 255;
    buf[ci][w] = Qraw[((size_t)(b * C0 + c0 + ci) * H + h) * W + w];
  }
  __syncthreads();
  if (t < 32) { float s = 0.f; for (int w = 0; w < W; ++w) s += buf[t][w]; mean[t] = s * (1.f / (float)W); }
  __syncthreads();
  for (int i = t; i < 32 * W; i += 256) {
    int w = i >> 5, ci = i & 31;
    Qt[(bh * W + w) * 64 + c0 + ci] = (__bf16)(buf[ci][w] - mean[ci]);
  }
}

// score S[b,h,u,v] = sum_c Qt[b,h,u,c] Kt[b,h,v,c]
__global__ void k_score(const __bf16* __restrict__ Qt, const __bf16* __restrict__ Kt,
                        float* __restrict__ S) {
  const int lane = threadIdx.x;
  const int ut = blockIdx.x & 15, vt = blockIdx.x >> 4;
  const int h = blockIdx.y, b = blockIdx.z;
  const size_t bh = (size_t)b * H + h;
  const int m = lane & 15, kd = lane >> 4;
  const __bf16* Qrow = Qt + (bh * W + ut * 16 + m) * 64;
  const __bf16* Krow = Kt + (bh * W + vt * 16 + m) * 64;
  floatx8 acc = {};
#pragma unroll
  for (int ks = 0; ks < 2; ++ks) {
    bf16x8 lo = *(const bf16x8*)(Qrow + ks * 32 + (kd << 4));
    bf16x8 hi = *(const bf16x8*)(Qrow + ks * 32 + 8 + (kd << 4));
    bf16x16 a;
#pragma unroll
    for (int e = 0; e < 8; ++e) { a[e] = lo[e]; a[e + 8] = hi[e]; }
    bf16x16 bb = *(const bf16x16*)(Krow + ks * 32 + (kd << 4));
    acc = wmma_bf16(a, bb, acc);
  }
#pragma unroll
  for (int r = 0; r < 8; ++r) {
    int uu = ut * 16 + r + (kd << 3);
    S[(bh * W + uu) * W + vt * 16 + m] = acc[r];
  }
}

// row softmax (over v), bf16 out
__global__ void k_rowsm(const float* __restrict__ S, __bf16* __restrict__ RSb) {
  __shared__ float red[256];
  int t = threadIdx.x;
  size_t row = blockIdx.x;
  float v = S[row * W + t];
  red[t] = v; __syncthreads();
  for (int s = 128; s > 0; s >>= 1) { if (t < s) red[t] = fmaxf(red[t], red[t + s]); __syncthreads(); }
  float mx = red[0]; __syncthreads();
  float e = __expf(v - mx);
  red[t] = e; __syncthreads();
  for (int s = 128; s > 0; s >>= 1) { if (t < s) red[t] += red[t + s]; __syncthreads(); }
  RSb[row * W + t] = (__bf16)(e / red[0]);
}

// column softmax (over u): CSb in S layout (for gates) + ML row-major (M_l2r[w,v], for warp GEMM)
__global__ void k_colsm(const float* __restrict__ S, __bf16* __restrict__ CSb,
                        __bf16* __restrict__ ML) {
  int v = threadIdx.x;
  size_t bh = blockIdx.x;
  const float* Sp = S + bh * (size_t)W * W;
  float mx = -1e30f;
  for (int u = 0; u < W; ++u) mx = fmaxf(mx, Sp[(size_t)u * W + v]);
  float sum = 0.f;
  for (int u = 0; u < W; ++u) sum += __expf(Sp[(size_t)u * W + v] - mx);
  float inv = 1.f / sum;
  __bf16* Cp = CSb + bh * (size_t)W * W;
  __bf16* Mp = ML + bh * (size_t)W * W;
  for (int u = 0; u < W; ++u) {
    float val = __expf(Sp[(size_t)u * W + v] - mx) * inv;
    Cp[(size_t)u * W + v] = (__bf16)val;       // CS[u,v]
    Mp[(size_t)v * W + u] = (__bf16)val;       // ML[v,u] = M_l2r[v,u] (thread-contiguous row)
  }
}

// gL[b,h,u] = tanh(5 * sum_v (sum_k RS[u+k,v]) * CS[u,v])
__global__ void k_vleft(const __bf16* __restrict__ RSb, const __bf16* __restrict__ CSb,
                        float* __restrict__ gL) {
  __shared__ float red[256];
  int t = threadIdx.x;
  size_t row = blockIdx.x;
  size_t bh = row / W; int u = (int)(row % W);
  const __bf16* R = RSb + bh * (size_t)W * W;
  const __bf16* Cc = CSb + bh * (size_t)W * W;
  float rs = 0.f;
#pragma unroll
  for (int k = -2; k <= 2; ++k) {
    int uu = u + k;
    float mk = (uu >= 0 && uu < W) ? 1.f : 0.f;
    int uc = uu < 0 ? 0 : (uu >= W ? W - 1 : uu);
    rs += mk * (float)R[(size_t)uc * W + t];
  }
  red[t] = rs * (float)Cc[(size_t)u * W + t];
  __syncthreads();
  for (int s = 128; s > 0; s >>= 1) { if (t < s) red[t] += red[t + s]; __syncthreads(); }
  if (t == 0) gL[row] = tanhf(5.f * red[0]);
}

// gR[b,h,u] = tanh(5 * sum_v (sum_k CS[v,u+k]) * RS[v,u])
__global__ void k_vright(const __bf16* __restrict__ RSb, const __bf16* __restrict__ CSb,
                         float* __restrict__ gR) {
  int u = threadIdx.x;
  size_t bh = blockIdx.x;
  const __bf16* R = RSb + bh * (size_t)W * W;
  const __bf16* Cc = CSb + bh * (size_t)W * W;
  float acc = 0.f;
  for (int v = 0; v < W; ++v) {
    float cs = 0.f;
#pragma unroll
    for (int k = -2; k <= 2; ++k) {
      int uu = u + k;
      float mk = (uu >= 0 && uu < W) ? 1.f : 0.f;
      int uc = uu < 0 ? 0 : (uu >= W ? W - 1 : uu);
      cs += mk * (float)Cc[(size_t)v * W + uc];
    }
    acc += cs * (float)R[(size_t)v * W + u];
  }
  gR[bh * W + u] = tanhf(5.f * acc);
}

// warp+blend: out[b,c,h,w] = xsame*(1-g) + (sum_v Mrow[w,v]*xother[c,v]) * g
// grid (16 wtiles, 4ctiles*H, B), block 32. Mrow row-major [w][v] (RSb for left, ML for right).
__global__ void k_xt(const __bf16* __restrict__ Mrow, const __bf16* __restrict__ xob,
                     const float* __restrict__ xsame, const float* __restrict__ gv,
                     float* __restrict__ outp) {
  const int lane = threadIdx.x;
  const int wb = blockIdx.x * 16;
  const int ct = blockIdx.y & 3, h = blockIdx.y >> 2;
  const int b = blockIdx.z;
  const size_t bh = (size_t)b * H + h;
  const int m = lane & 15, kd = lane >> 4;
  const __bf16* arow = xob + ((size_t)(b * C0 + ct * 16 + m) * H + h) * W;
  const __bf16* brow = Mrow + bh * (size_t)W * W + (size_t)(wb + m) * W;
  floatx8 acc = {};
#pragma unroll
  for (int vs = 0; vs < 8; ++vs) {
    bf16x8 lo = *(const bf16x8*)(arow + vs * 32 + (kd << 4));
    bf16x8 hi = *(const bf16x8*)(arow + vs * 32 + 8 + (kd << 4));
    bf16x16 a;
#pragma unroll
    for (int e = 0; e < 8; ++e) { a[e] = lo[e]; a[e + 8] = hi[e]; }
    bf16x16 bb = *(const bf16x16*)(brow + vs * 32 + (kd << 4));
    acc = wmma_bf16(a, bb, acc);
  }
#pragma unroll
  for (int r = 0; r < 8; ++r) {
    int c = ct * 16 + r + (kd << 3);
    int ww = wb + (lane & 15);
    float g = gv[bh * W + ww];
    size_t oi = ((size_t)(b * C0 + c) * H + h) * W + ww;
    outp[oi] = xsame[oi] * (1.f - g) + acc[r] * g;
  }
}

// ---------------- host launcher ----------------

extern "C" void kernel_launch(void* const* d_in, const int* in_sizes, int n_in,
                              void* d_out, int out_size, void* d_ws, size_t ws_size,
                              hipStream_t stream) {
  (void)in_sizes; (void)n_in; (void)out_size; (void)ws_size;
  const float* x_left  = (const float*)d_in[0];
  const float* x_right = (const float*)d_in[1];
  const float* cat[2]  = {(const float*)d_in[2], (const float*)d_in[3]};
  const float *bng = (const float*)d_in[4], *bnb = (const float*)d_in[5];
  const float *bnm = (const float*)d_in[6], *bnv = (const float*)d_in[7];
  const float* skw = (const float*)d_in[8];
  const float *skg = (const float*)d_in[9], *skbb = (const float*)d_in[10];
  const float *skm = (const float*)d_in[11], *skv = (const float*)d_in[12];
  const float* fc1w = (const float*)d_in[13];
  const float *f1g = (const float*)d_in[14], *f1bp = (const float*)d_in[15];
  const float *f1m = (const float*)d_in[16], *f1v = (const float*)d_in[17];
  const float* fc2w = (const float*)d_in[18];
  const float *bqw = (const float*)d_in[19], *bqb = (const float*)d_in[20];
  const float *bsw = (const float*)d_in[21], *bsb = (const float*)d_in[22];

  char* ws = (char*)d_ws;
  constexpr size_t SZ_X0B   = (size_t)Bsz * C0 * HW * 2;
  constexpr size_t OFF_XLB  = 0;
  constexpr size_t OFF_XRB  = OFF_XLB + SZ_X0B;
  constexpr size_t OFF_QT   = OFF_XRB + SZ_X0B;
  constexpr size_t OFF_KT   = OFF_QT + SZ_X0B;
  constexpr size_t OFF_POOL = OFF_KT + SZ_X0B;
  constexpr size_t OFF_AW   = OFF_POOL + 4096;
  constexpr size_t OFF_BNS  = OFF_AW + 16384;
  constexpr size_t OFF_BNB  = OFF_BNS + 1024;
  constexpr size_t OFF_SKS  = OFF_BNB + 1024;
  constexpr size_t OFF_SKB  = OFF_SKS + 4096;
  constexpr size_t OFF_F1S  = OFF_SKB + 4096;
  constexpr size_t OFF_F1B  = OFF_F1S + 256;
  constexpr size_t OFF_WP   = OFF_F1B + 256;                     // 4*16*5*32*16*2 = 327680
  constexpr size_t OFF_QWP  = OFF_WP + 327680;                   // 16*32*16*2 = 16384
  constexpr size_t OFF_GL   = OFF_QWP + 16384;
  constexpr size_t OFF_GR   = OFF_GL + (size_t)Bsz * H * W * 4;
  constexpr size_t OFF_SH   = OFF_GR + (size_t)Bsz * H * W * 4;
  // conv-phase aliases
  constexpr size_t OFF_XB2  = OFF_SH;
  constexpr size_t OFF_OUTS = OFF_XB2 + (size_t)Bsz * C * HW * 2;
  constexpr size_t OFF_VB   = OFF_OUTS + (size_t)Mbr * Bsz * C * HW * 2;
  constexpr size_t OFF_QRAW = OFF_VB + (size_t)Bsz * C * HW * 2;
  // attn-phase aliases (conv-phase buffers dead by then)
  constexpr size_t OFF_S    = OFF_SH;
  constexpr size_t OFF_RSB  = OFF_S + (size_t)Bsz * H * W * W * 4;
  constexpr size_t OFF_CSB  = OFF_RSB + (size_t)Bsz * H * W * W * 2;
  constexpr size_t OFF_ML   = OFF_CSB + (size_t)Bsz * H * W * W * 2;

  __bf16* xlb = (__bf16*)(ws + OFF_XLB);
  __bf16* xrb = (__bf16*)(ws + OFF_XRB);
  __bf16* Qt  = (__bf16*)(ws + OFF_QT);
  __bf16* Kt  = (__bf16*)(ws + OFF_KT);
  float*  pool = (float*)(ws + OFF_POOL);
  float*  aw   = (float*)(ws + OFF_AW);
  float *BNS = (float*)(ws + OFF_BNS), *BNB = (float*)(ws + OFF_BNB);
  float *SKS = (float*)(ws + OFF_SKS), *SKB = (float*)(ws + OFF_SKB);
  float *F1S = (float*)(ws + OFF_F1S), *F1B = (float*)(ws + OFF_F1B);
  __bf16* wp  = (__bf16*)(ws + OFF_WP);
  __bf16* qwp = (__bf16*)(ws + OFF_QWP);
  float *gL = (float*)(ws + OFF_GL), *gR = (float*)(ws + OFF_GR);
  __bf16* xb2  = (__bf16*)(ws + OFF_XB2);
  __bf16* outs = (__bf16*)(ws + OFF_OUTS);
  __bf16* Vb   = (__bf16*)(ws + OFF_VB);
  float*  Qraw = (float*)(ws + OFF_QRAW);
  float*  S    = (float*)(ws + OFF_S);
  __bf16* RSb  = (__bf16*)(ws + OFF_RSB);
  __bf16* CSb  = (__bf16*)(ws + OFF_CSB);
  __bf16* ML   = (__bf16*)(ws + OFF_ML);

  float* outL = (float*)d_out;
  float* outR = outL + (size_t)Bsz * C0 * HW;

  k_fold<<<4, 256, 0, stream>>>(bng, bnb, bnm, bnv, skg, skbb, skm, skv,
                                f1g, f1bp, f1m, f1v, BNS, BNB, SKS, SKB, F1S, F1B);
  k_packw<<<320, 32, 0, stream>>>(skw, wp);
  k_packqw<<<16, 32, 0, stream>>>(bqw, bsw, qwp);
  const size_t nx0 = (size_t)Bsz * C0 * HW;
  k_tobf<<<(unsigned)(nx0 / 256), 256, 0, stream>>>(x_left, xlb, nx0);
  k_tobf<<<(unsigned)(nx0 / 256), 256, 0, stream>>>(x_right, xrb, nx0);

  for (int side = 0; side < 2; ++side) {
    k_bncat<<<dim3(HW / 16, 16, Bsz), 256, 0, stream>>>(cat[side], BNS, BNB, xb2);
    k_skconv<<<dim3(W / 16, H, Bsz * Mbr * 16), 32, 0, stream>>>(xb2, wp, SKS, SKB, outs);
    k_pool<<<Bsz * C, 256, 0, stream>>>(outs, pool);
    k_skattn<<<1, 256, 0, stream>>>(pool, fc1w, F1S, F1B, fc2w, aw);
    k_wv<<<dim3(HW / 16, 16, Bsz), 256, 0, stream>>>(outs, aw, Vb);
    k_qconv<<<dim3(W / 16, H, Bsz * 4), 32, 0, stream>>>(Vb, qwp, side ? bsb : bqb, Qraw, side);
    k_meansub<<<dim3(Bsz * H, 2), 256, 0, stream>>>(Qraw, side ? Kt : Qt);
  }

  k_score<<<dim3(256, H, Bsz), 32, 0, stream>>>(Qt, Kt, S);
  k_rowsm<<<Bsz * H * W, 256, 0, stream>>>(S, RSb);
  k_colsm<<<Bsz * H, 256, 0, stream>>>(S, CSb, ML);
  k_vleft<<<Bsz * H * W, 256, 0, stream>>>(RSb, CSb, gL);
  k_vright<<<Bsz * H, 256, 0, stream>>>(RSb, CSb, gR);
  k_xt<<<dim3(16, 4 * H, Bsz), 32, 0, stream>>>(RSb, xrb, x_left, gL, outL);
  k_xt<<<dim3(16, 4 * H, Bsz), 32, 0, stream>>>(ML, xlb, x_right, gR, outR);
}
